// EfficientMultiScaleROIExtractor_4818953306514
// MI455X (gfx1250) — matrix operand order repaired and checked by reading, compiled, not run on hardware
//
#include <hip/hip_runtime.h>
#include <hip/hip_bf16.h>
#include <stdint.h>

// ---------------------------------------------------------------------------
// EfficientMultiScaleROIExtractor for MI455X (gfx1250, wave32, WMMA)
//
// Pipeline:
//   1) transpose fm [64,120,120] -> fmT [14400,64]  (coalesced ROI gathers)
//   2) per-channel global mean -> g[64]; head(g) once -> gfeat[64] (bf16)
//   3) pre-format weights to bf16 in WMMA-B register layout
//   4) ROI bilinear pool (3 scales) -> pooled bf16 [3,4096,64]
//   5) bf16 WMMA GEMMs (fp32 accum, compile-time K, 2 M-tiles/wave):
//        per scale: pooled@w1 -> H1[4096,128]; H1@w2 -> concat[:, s*64:+64]
//        broadcast gfeat      -> concat[:, 192:256]
//        concat@p1 -> H[4096,128]; H@p2 -> out fp32 [4096,64]
// ---------------------------------------------------------------------------

typedef __attribute__((ext_vector_type(16))) __bf16 v16bf;
typedef __attribute__((ext_vector_type(8)))  float  v8f;

#define FM_C 64
#define FM_H 120
#define FM_W 120
#define FM_HW (FM_H * FM_W)   // 14400
#define NBOX 4096
#define IMG 960.0f

__device__ __forceinline__ unsigned short f2bf(float f) {
    union { float f; unsigned u; } x; x.f = f;
    unsigned u = x.u;
    if ((u & 0x7fffffffu) > 0x7f800000u) return 0x7fc0;           // NaN
    unsigned r = (u + 0x7fffu + ((u >> 16) & 1u)) >> 16;          // RNE
    return (unsigned short)r;
}

// --------------------------- 1) transpose fm -------------------------------
// fm [C=64][HW=14400] -> fmT [HW][C], LDS tiled 64sp x 64ch
__global__ void k_transpose_fm(const float* __restrict__ fm, float* __restrict__ fmT) {
    __shared__ float t[64][65];
    const int sp0 = blockIdx.x * 64;               // 225 blocks
    const int tid = threadIdx.x;                   // 256 threads
    #pragma unroll
    for (int rep = 0; rep < 16; ++rep) {
        int ch = rep * 4 + (tid >> 6);
        int sp = tid & 63;
        t[sp][ch] = fm[ch * FM_HW + sp0 + sp];     // contiguous along sp
    }
    __syncthreads();
    #pragma unroll
    for (int rep = 0; rep < 16; ++rep) {
        int ch = tid & 63;
        int sp = rep * 4 + (tid >> 6);
        fmT[(size_t)(sp0 + sp) * 64 + ch] = t[sp][ch];  // contiguous along ch
    }
}

// --------------------------- 2a) global mean -------------------------------
__global__ void k_global_mean(const float* __restrict__ fm, float* __restrict__ g) {
    __shared__ float red[256];
    const int c = blockIdx.x;                      // 64 blocks
    const int tid = threadIdx.x;                   // 256 threads
    float s = 0.0f;
    for (int i = tid; i < FM_HW; i += 256) s += fm[c * FM_HW + i];
    red[tid] = s;
    __syncthreads();
    for (int ofs = 128; ofs > 0; ofs >>= 1) {
        if (tid < ofs) red[tid] += red[tid + ofs];
        __syncthreads();
    }
    if (tid == 0) g[c] = red[0] * (1.0f / (float)FM_HW);
}

// --------------------------- 2b) head(global) once -------------------------
__global__ void k_gfeat(const float* __restrict__ g,
                        const float* __restrict__ w1, const float* __restrict__ b1,
                        const float* __restrict__ w2, const float* __restrict__ b2,
                        unsigned short* __restrict__ gfeat) {
    __shared__ float gs[64];
    __shared__ float h1[128];
    const int tid = threadIdx.x;                   // 128 threads
    if (tid < 64) gs[tid] = g[tid];
    __syncthreads();
    float acc = b1[tid];
    for (int c = 0; c < 64; ++c) acc += gs[c] * w1[c * 128 + tid];
    h1[tid] = fmaxf(acc, 0.0f);
    __syncthreads();
    if (tid < 64) {
        float a2 = b2[tid];
        for (int k = 0; k < 128; ++k) a2 += h1[k] * w2[k * 64 + tid];
        gfeat[tid] = f2bf(fmaxf(a2, 0.0f));
    }
}

// --------------------------- 3) pack weights to WMMA-B layout --------------
// B [K x N] fp32 row-major -> packed dwords:
//   dst[((kstep*(N/16) + ntile)*32 + lane)*8 + v] = { B[kb,n], B[kb+1,n] } bf16
//   with n = ntile*16 + lane%16, half = lane/16, kb = kstep*32 + half*16 + v*2
__global__ void k_pack_b(const float* __restrict__ src, unsigned* __restrict__ dst,
                         int K, int N) {
    const int total = (K * N) >> 1;
    int i = blockIdx.x * blockDim.x + threadIdx.x;
    if (i >= total) return;
    const int v    = i & 7;
    const int lane = (i >> 3) & 31;
    const int tile = i >> 8;
    const int ntiles = N >> 4;
    const int ntile = tile % ntiles;
    const int kstep = tile / ntiles;
    const int n  = ntile * 16 + (lane & 15);
    const int kb = kstep * 32 + (lane >> 4) * 16 + v * 2;
    unsigned lo = f2bf(src[(size_t)kb * N + n]);
    unsigned hi = f2bf(src[(size_t)(kb + 1) * N + n]);
    dst[i] = lo | (hi << 16);
}

// --------------------------- 4) ROI bilinear pool --------------------------
// one block (64 threads = channels) per box; all 3 scales.
__global__ void k_roi_pool(const float* __restrict__ fmT,
                           const float* __restrict__ boxes,
                           unsigned short* __restrict__ pooled /* [3][N][64] */) {
    const int n = blockIdx.x;
    const int t = threadIdx.x;                     // channel
    const float k2 = 2.0f / IMG;
    const float x1 = boxes[n * 4 + 0] * k2 - 1.0f;
    const float y1 = boxes[n * 4 + 1] * k2 - 1.0f;
    const float x2 = boxes[n * 4 + 2] * k2 - 1.0f;
    const float y2 = boxes[n * 4 + 3] * k2 - 1.0f;
    const float cx = 0.5f * (x1 + x2);
    const float cy = 0.5f * (y1 + y2);
    const float hw = 0.5f * fmaxf(x2 - x1, 1e-6f);
    const float hh = 0.5f * fmaxf(y2 - y1, 1e-6f);

    const int rs[3] = {3, 7, 11};
    for (int s = 0; s < 3; ++s) {
        const int r = rs[s];
        const float invr = 1.0f / (float)r;
        float sum = 0.0f;
        for (int py = 0; py < r; ++py) {
            const float liny = (2.0f * py + 1.0f) * invr - 1.0f;
            const float gy = cy + hh * liny;
            const float iy = ((gy + 1.0f) * (float)FM_H - 1.0f) * 0.5f;
            const float y0f = floorf(iy);
            const float wy1 = iy - y0f;
            const int   y0  = (int)y0f;
            const bool  vy0 = (y0 >= 0) && (y0 < FM_H);
            const bool  vy1 = (y0 + 1 >= 0) && (y0 + 1 < FM_H);
            const int   y0c = min(max(y0, 0), FM_H - 1);
            const int   y1c = min(max(y0 + 1, 0), FM_H - 1);
            for (int px = 0; px < r; ++px) {
                const float linx = (2.0f * px + 1.0f) * invr - 1.0f;
                const float gx = cx + hw * linx;
                const float ix = ((gx + 1.0f) * (float)FM_W - 1.0f) * 0.5f;
                const float x0f = floorf(ix);
                const float wx1 = ix - x0f;
                const int   x0  = (int)x0f;
                const bool  vx0 = (x0 >= 0) && (x0 < FM_W);
                const bool  vx1 = (x0 + 1 >= 0) && (x0 + 1 < FM_W);
                const int   x0c = min(max(x0, 0), FM_W - 1);
                const int   x1c = min(max(x0 + 1, 0), FM_W - 1);
                const float w00 = (1.0f - wx1) * (1.0f - wy1) * ((vx0 && vy0) ? 1.0f : 0.0f);
                const float w10 = wx1 * (1.0f - wy1)          * ((vx1 && vy0) ? 1.0f : 0.0f);
                const float w01 = (1.0f - wx1) * wy1          * ((vx0 && vy1) ? 1.0f : 0.0f);
                const float w11 = wx1 * wy1                   * ((vx1 && vy1) ? 1.0f : 0.0f);
                const float v00 = fmT[(size_t)(y0c * FM_W + x0c) * 64 + t];
                const float v10 = fmT[(size_t)(y0c * FM_W + x1c) * 64 + t];
                const float v01 = fmT[(size_t)(y1c * FM_W + x0c) * 64 + t];
                const float v11 = fmT[(size_t)(y1c * FM_W + x1c) * 64 + t];
                sum = fmaf(w00, v00, fmaf(w10, v10, fmaf(w01, v01, fmaf(w11, v11, sum))));
            }
        }
        pooled[((size_t)s * NBOX + n) * 64 + t] = f2bf(sum / (float)(r * r));
    }
}

// --------------------------- 4b) broadcast gfeat into concat ---------------
__global__ void k_fill_concat(const unsigned short* __restrict__ gfeat,
                              unsigned short* __restrict__ concat /* [N][256] */) {
    int i = blockIdx.x * blockDim.x + threadIdx.x;
    if (i >= NBOX * 64) return;
    int n = i >> 6, c = i & 63;
    concat[(size_t)n * 256 + 192 + c] = gfeat[c];
}

// --------------------------- 5) bf16 WMMA GEMM + bias + ReLU ---------------
// A: bf16 row-major [M x K], K = KSTEPS*32 (compile-time -> full unroll,
// loads pipelined across WMMAs). Bf: pre-packed WMMA-B dwords. One wave
// computes TWO vertically adjacent 16x16 output tiles sharing one B panel:
// each B register pair feeds two back-to-back v_wmma_f32_16x16x32_bf16.
template <int KSTEPS, bool OUT_F32>
__global__ void k_gemm_wmma(const unsigned short* __restrict__ A,
                            const unsigned* __restrict__ Bf,
                            const float* __restrict__ bias,
                            void* __restrict__ outp,
                            int M, int N, int out_stride, int col_offset) {
    constexpr int K = KSTEPS * 32;
    const int lane = threadIdx.x & 31;
    const int wave = threadIdx.x >> 5;
    const int tiles_n = N >> 4;
    const int tile = blockIdx.x * 4 + wave;         // pair-tile id
    const int tile_mp = tile / tiles_n;             // M-pair index (32 rows)
    const int tile_n  = tile % tiles_n;
    if (tile_mp >= (M >> 5)) return;                // wave-uniform

    const int half = lane >> 4;
    const int lm   = lane & 15;
    const unsigned short* arow0 = A + (size_t)(tile_mp * 32 + lm) * K;
    const unsigned short* arow1 = arow0 + (size_t)16 * K;

    v8f c0 = {};
    v8f c1 = {};
    #pragma unroll
    for (int ks = 0; ks < KSTEPS; ++ks) {
        union { v16bf v; unsigned u[8]; } a0, a1, b;
        const int kb = ks * 32 + half * 16;
        #pragma unroll
        for (int v = 0; v < 8; ++v) {
            const int kp = kb + v * 2;   // contiguous k = 16*half + s, matches packed B
            a0.u[v] = *(const unsigned*)(arow0 + kp);
            a1.u[v] = *(const unsigned*)(arow1 + kp);
        }
        const unsigned* bp = Bf + ((size_t)(ks * tiles_n + tile_n) * 32 + lane) * 8;
        #pragma unroll
        for (int v = 0; v < 8; ++v) b.u[v] = bp[v];
        c0 = __builtin_amdgcn_wmma_f32_16x16x32_bf16(
                false, a0.v, false, b.v, (short)0, c0, false, false);
        c1 = __builtin_amdgcn_wmma_f32_16x16x32_bf16(
                false, a1.v, false, b.v, (short)0, c1, false, false);
    }

    const int col = tile_n * 16 + lm;
    const float bv = bias[col];
    #pragma unroll
    for (int v = 0; v < 8; ++v) {
        const int row0 = tile_mp * 32 + v + half * 8;               // ISA C/D layout
        const int row1 = row0 + 16;
        float val0 = fmaxf(c0[v] + bv, 0.0f);
        float val1 = fmaxf(c1[v] + bv, 0.0f);
        if (OUT_F32) {
            ((float*)outp)[(size_t)row0 * out_stride + col_offset + col] = val0;
            ((float*)outp)[(size_t)row1 * out_stride + col_offset + col] = val1;
        } else {
            ((unsigned short*)outp)[(size_t)row0 * out_stride + col_offset + col] = f2bf(val0);
            ((unsigned short*)outp)[(size_t)row1 * out_stride + col_offset + col] = f2bf(val1);
        }
    }
}

// ---------------------------------------------------------------------------
extern "C" void kernel_launch(void* const* d_in, const int* in_sizes, int n_in,
                              void* d_out, int out_size, void* d_ws, size_t ws_size,
                              hipStream_t stream) {
    const float* fm    = (const float*)d_in[0];
    const float* boxes = (const float*)d_in[1];
    const float* w1    = (const float*)d_in[2];
    const float* b1    = (const float*)d_in[3];
    const float* w2    = (const float*)d_in[4];
    const float* b2    = (const float*)d_in[5];
    const float* p1    = (const float*)d_in[6];
    const float* pb1   = (const float*)d_in[7];
    const float* p2    = (const float*)d_in[8];
    const float* pb2   = (const float*)d_in[9];
    float* out = (float*)d_out;

    char* w = (char*)d_ws;
    float*          fmT    = (float*)(w);                               // 3,686,400 B
    float*          g      = (float*)(w + 3686400);                     //       256 B
    unsigned short* gfeat  = (unsigned short*)(w + 3686656);            //       128 B
    unsigned short* pooled = (unsigned short*)(w + 3686784);            // 1,572,864 B
    unsigned short* H1     = (unsigned short*)(w + 5259648);            // 1,048,576 B
    unsigned short* concat = (unsigned short*)(w + 6308224);            // 2,097,152 B
    unsigned short* H2     = (unsigned short*)(w + 8405376);            // 1,048,576 B
    unsigned*       w1f    = (unsigned*)(w + 9453952);                  //    16,384 B
    unsigned*       w2f    = (unsigned*)(w + 9470336);                  //    16,384 B
    unsigned*       p1f    = (unsigned*)(w + 9486720);                  //    65,536 B
    unsigned*       p2f    = (unsigned*)(w + 9552256);                  //    16,384 B

    // prep
    k_transpose_fm<<<FM_HW / 64, 256, 0, stream>>>(fm, fmT);
    k_global_mean<<<64, 256, 0, stream>>>(fm, g);
    k_gfeat<<<1, 128, 0, stream>>>(g, w1, b1, w2, b2, gfeat);
    k_pack_b<<<16, 256, 0, stream>>>(w1, w1f, 64, 128);
    k_pack_b<<<16, 256, 0, stream>>>(w2, w2f, 128, 64);
    k_pack_b<<<64, 256, 0, stream>>>(p1, p1f, 256, 128);
    k_pack_b<<<16, 256, 0, stream>>>(p2, p2f, 128, 64);

    // ROI pooling, 3 scales
    k_roi_pool<<<NBOX, 64, 0, stream>>>(fmT, boxes, pooled);

    // broadcast global-feature head output into concat[:,192:256]
    k_fill_concat<<<(NBOX * 64) / 256, 256, 0, stream>>>(gfeat, concat);

    // per-scale shared head: pooled(4096x64) @ w1 -> H1(4096x128) @ w2 -> concat
    for (int s = 0; s < 3; ++s) {
        const unsigned short* As = pooled + (size_t)s * NBOX * 64;
        // GEMM1: M=4096 N=128 K=64  -> (4096/32)*(128/16)=1024 pair-tiles -> 256 blocks
        k_gemm_wmma<2, false><<<256, 128, 0, stream>>>(As, w1f, b1, H1,
                                                       NBOX, 128, 128, 0);
        // GEMM2: M=4096 N=64 K=128 -> 512 pair-tiles -> 128 blocks, into concat
        k_gemm_wmma<4, false><<<128, 128, 0, stream>>>(H1, w2f, b2, concat,
                                                       NBOX, 64, 256, s * 64);
    }

    // final head: concat(4096x256) @ p1 -> H2(4096x128) @ p2 -> out fp32(4096x64)
    k_gemm_wmma<8, false><<<256, 128, 0, stream>>>(concat, p1f, pb1, H2,
                                                   NBOX, 128, 128, 0);
    k_gemm_wmma<4, true><<<128, 128, 0, stream>>>(H2, p2f, pb2, out,
                                                  NBOX, 64, 64, 0);
}
